// Qwen3_5MoeGatedDeltaNet_72670846649121
// MI455X (gfx1250) — matrix operand order repaired and checked by reading, not compiled, not run on hardware
//
#include <hip/hip_runtime.h>
#include <hip/hip_bf16.h>
#include <math.h>

// ---------------- problem constants ----------------
#define T_TOKENS   4096
#define HIDDEN     2048
#define NUM_K_HEADS 16
#define NUM_V_HEADS 32
#define DK         128
#define DV         128
#define KEY_DIM    (NUM_K_HEADS * DK)          // 2048
#define VALUE_DIM  (NUM_V_HEADS * DV)          // 4096
#define QKV_DIM    (2 * KEY_DIM + VALUE_DIM)   // 8192
#define KCONV      4
#define CHUNK      32                          // delta-rule chunk (== WMMA K)
#define LDT        132                         // tile leading dim (floats)
#define LDA        33                          // 32x32 matrix leading dim

typedef __attribute__((ext_vector_type(16))) __bf16 v16bf;
typedef __attribute__((ext_vector_type(8)))  float  v8f;

union FragBF16 {
    v16bf v;
    uint4 q[2];
};

// =====================================================================
// Async global->LDS copy (gfx1250): 16 bytes per lane, tracked on
// ASYNCcnt. VDST VGPR carries the LDS byte address, VADDR the 64-bit
// global address (GV mode). Cross-wave visibility = s_wait_asynccnt 0
// in each wave followed by a workgroup barrier.
// =====================================================================
__device__ __forceinline__ void async_copy16(const float* gptr, const float* lptr)
{
    const unsigned lds = (unsigned)(unsigned long long)(const void*)lptr;
    asm volatile("global_load_async_to_lds_b128 %0, %1, off"
                 :
                 : "v"(lds), "v"((unsigned long long)gptr)
                 : "memory");
}
#define ASYNC_WAIT0() asm volatile("s_wait_asynccnt 0x0" ::: "memory")

// =====================================================================
// Fragment builders: read f32 LDS, convert to bf16 in-register.
// Layouts per ISA 7.12.2 (wave32):
//  A 16x32: lane m=lane&15, g=lane>>4; halfs[0..7]=K(8g..8g+7),
//           halfs[8..15]=K(8g+16..8g+23)
//  B 32x16: lane n=lane&15, g=lane>>4; halfs[0..7]=K(8g..8g+7),
//           halfs[8..15]=K(8g+16..8g+23)
//  C/D:     VGPR i -> row = 8g+i, col = lane&15
// =====================================================================
__device__ __forceinline__ v16bf fragA_f32(const float* p, int ld, int m0, int k0, int lane)
{
    const int ln = lane & 15, g = lane >> 4;
    const float* r = p + (m0 + ln) * ld + k0 + 8 * g;
    v16bf f;
#pragma unroll
    for (int i = 0; i < 8; ++i) f[i]     = (__bf16)r[i];
#pragma unroll
    for (int i = 0; i < 8; ++i) f[8 + i] = (__bf16)r[16 + i];
    return f;
}

// B stored K-major: p[k][n], leading dim ld
__device__ __forceinline__ v16bf fragB_kmajor_f32(const float* p, int ld, int k0, int n0, int lane)
{
    const int ln = lane & 15, g = lane >> 4;
    const float* c = p + (k0 + 8 * g) * ld + n0 + ln;
    v16bf f;
#pragma unroll
    for (int i = 0; i < 8; ++i) f[i] = (__bf16)c[i * ld];
#pragma unroll
    for (int i = 0; i < 8; ++i) f[8 + i] = (__bf16)c[(16 + i) * ld];
    return f;
}

// B stored N-major: p[n][k] (i.e. we multiply by p^T), leading dim ld
__device__ __forceinline__ v16bf fragB_nmajor_f32(const float* p, int ld, int k0, int n0, int lane)
{
    const int ln = lane & 15, g = lane >> 4;
    const float* r = p + (n0 + ln) * ld + k0 + 8 * g;
    v16bf f;
#pragma unroll
    for (int i = 0; i < 8; ++i) f[i] = (__bf16)r[i];
#pragma unroll
    for (int i = 0; i < 8; ++i) f[8 + i] = (__bf16)r[16 + i];
    return f;
}

// A = (p^T) with per-k scale: A[m][k] = p[k][m0+m] * dv[k]   (K=32 tile)
__device__ __forceinline__ v16bf fragAT_scaled_f32(const float* p, int ld, int m0,
                                                   const float* dv, int lane)
{
    const int ln = lane & 15, g = lane >> 4;
    const int m = m0 + ln;
    v16bf f;
#pragma unroll
    for (int i = 0; i < 8; ++i) { const int kk = 8 * g + i;      f[i]     = (__bf16)(p[kk * ld + m] * dv[kk]); }
#pragma unroll
    for (int i = 0; i < 8; ++i) { const int kk = 16 + 8 * g + i; f[8 + i] = (__bf16)(p[kk * ld + m] * dv[kk]); }
    return f;
}

__device__ __forceinline__ void storeAcc(float* p, int ld, int m0, int n0, int lane, v8f acc)
{
    const int ln = lane & 15, g = lane >> 4;
#pragma unroll
    for (int i = 0; i < 8; ++i) p[(m0 + 8 * g + i) * ld + n0 + ln] = acc[i];
}

__device__ __forceinline__ void addAcc(float* p, int ld, int m0, int n0, int lane, v8f acc)
{
    const int ln = lane & 15, g = lane >> 4;
#pragma unroll
    for (int i = 0; i < 8; ++i) p[(m0 + 8 * g + i) * ld + n0 + ln] += acc[i];
}

#define WMMA_BF16(a, b, c) \
    __builtin_amdgcn_wmma_f32_16x16x32_bf16(false, (a), false, (b), (short)0, (c), false, false)

// =====================================================================
// WMMA GEMM:  C[M,N] = A[M,K] * B[N,K]^T   (A,B,C row-major f32)
// Block: 256 threads = 8 waves; block tile 64(M) x 32(N); K-step 32.
// (keeps VGPR staging: f32->bf16 conversion happens in flight)
// =====================================================================
__global__ __launch_bounds__(256)
void wmma_gemm_nt(const float* __restrict__ A, const float* __restrict__ B,
                  float* __restrict__ C, int M, int N, int K)
{
    __shared__ __attribute__((aligned(16))) __bf16 As[64][40];
    __shared__ __attribute__((aligned(16))) __bf16 Bs[32][40];

    const int tid  = threadIdx.x;
    const int lane = tid & 31;
    const int wave = tid >> 5;
    const int wm   = wave >> 1;
    const int wn   = wave & 1;
    const int g    = lane >> 4;
    const int ln   = lane & 15;

    const int m0 = blockIdx.y * 64;
    const int n0 = blockIdx.x * 32;

    v8f acc = {};

    for (int k0 = 0; k0 < K; k0 += 32) {
        {
            const int r = tid >> 2;
            const int c = (tid & 3) * 8;
            const float* ap = A + (size_t)(m0 + r) * K + k0 + c;
            float4 f0 = *(const float4*)ap;
            float4 f1 = *(const float4*)(ap + 4);
            As[r][c + 0] = (__bf16)f0.x;  As[r][c + 1] = (__bf16)f0.y;
            As[r][c + 2] = (__bf16)f0.z;  As[r][c + 3] = (__bf16)f0.w;
            As[r][c + 4] = (__bf16)f1.x;  As[r][c + 5] = (__bf16)f1.y;
            As[r][c + 6] = (__bf16)f1.z;  As[r][c + 7] = (__bf16)f1.w;
        }
        {
            const int r = tid >> 3;
            const int c = (tid & 7) * 4;
            const float* bp = B + (size_t)(n0 + r) * K + k0 + c;
            float4 f = *(const float4*)bp;
            Bs[r][c + 0] = (__bf16)f.x;  Bs[r][c + 1] = (__bf16)f.y;
            Bs[r][c + 2] = (__bf16)f.z;  Bs[r][c + 3] = (__bf16)f.w;
        }
        __syncthreads();

        FragBF16 a, b;
        a.q[0] = *(const uint4*)&As[wm * 16 + ln][8 * g];
        a.q[1] = *(const uint4*)&As[wm * 16 + ln][8 * g + 16];
        b.q[0] = *(const uint4*)&Bs[wn * 16 + ln][8 * g];
        b.q[1] = *(const uint4*)&Bs[wn * 16 + ln][8 * g + 16];

        acc = WMMA_BF16(a.v, b.v, acc);
        __syncthreads();
    }

#pragma unroll
    for (int i = 0; i < 8; ++i) {
        const int row = m0 + wm * 16 + g * 8 + i;
        const int col = n0 + wn * 16 + ln;
        C[(size_t)row * N + col] = acc[i];
    }
}

// =====================================================================
// Depthwise causal conv (k=4) + SiLU over mixed_qkv [T, QKV_DIM]
// =====================================================================
__global__ __launch_bounds__(256)
void conv_silu_kernel(const float* __restrict__ mixed,
                      const float* __restrict__ cw,
                      float* __restrict__ x)
{
    const long long idx = (long long)blockIdx.x * blockDim.x + threadIdx.x;
    if (idx >= (long long)T_TOKENS * QKV_DIM) return;
    const int t = (int)(idx / QKV_DIM);
    const int c = (int)(idx % QKV_DIM);
    float acc = 0.f;
#pragma unroll
    for (int i = 0; i < KCONV; ++i) {
        const int tt = t - (KCONV - 1) + i;
        if (tt >= 0) acc += mixed[(size_t)tt * QKV_DIM + c] * cw[c * KCONV + i];
    }
    x[idx] = acc / (1.f + __expf(-acc));
}

// =====================================================================
// L2 norm of q and k heads in place: rows = T * 32; one wave per row
// =====================================================================
__global__ __launch_bounds__(256)
void l2norm_qk_kernel(float* __restrict__ x)
{
    const int row  = blockIdx.x * 8 + (threadIdx.x >> 5);
    const int lane = threadIdx.x & 31;
    if (row >= T_TOKENS * 32) return;
    const int t = row >> 5;
    const int j = row & 31;
    float* p = x + (size_t)t * QKV_DIM + j * DK;

    float v[4]; float ss = 0.f;
#pragma unroll
    for (int i = 0; i < 4; ++i) { v[i] = p[lane + 32 * i]; ss += v[i] * v[i]; }
#pragma unroll
    for (int m = 16; m; m >>= 1) ss += __shfl_xor(ss, m, 32);
    const float r = rsqrtf(ss + 1e-6f);
#pragma unroll
    for (int i = 0; i < 4; ++i) p[lane + 32 * i] = v[i] * r;
}

// =====================================================================
// beta = sigmoid(b);  g = -exp(A_log) * softplus(a + dt_bias)
// =====================================================================
__global__ __launch_bounds__(256)
void beta_g_kernel(const float* __restrict__ b, const float* __restrict__ a,
                   const float* __restrict__ dt_bias, const float* __restrict__ A_log,
                   float* __restrict__ beta, float* __restrict__ g)
{
    const int idx = blockIdx.x * blockDim.x + threadIdx.x;
    if (idx >= T_TOKENS * NUM_V_HEADS) return;
    const int h = idx & (NUM_V_HEADS - 1);
    const float bb = b[idx];
    beta[idx] = 1.f / (1.f + __expf(-bb));
    const float u  = a[idx] + dt_bias[h];
    const float sp = (u > 20.f) ? u : log1pf(__expf(u));
    g[idx] = -__expf(A_log[h]) * sp;
}

// =====================================================================
// Chunked gated delta rule, one workgroup (256 thr, 8 waves) per v-head.
// State S[128][128] f32 persistent in LDS; per chunk of 32 tokens all
// dense work runs on v_wmma_f32_16x16x32_bf16. K/Q tile staging uses
// gfx1250 async global->LDS copies (ASYNCcnt).
// =====================================================================
__global__ __launch_bounds__(256)
void scan_chunked_kernel(const float* __restrict__ x,     // [T, QKV_DIM] post conv/norm
                         const float* __restrict__ beta,  // [T, 32]
                         const float* __restrict__ g,     // [T, 32]
                         float* __restrict__ o)           // [T, 32, 128]
{
    __shared__ __attribute__((aligned(16))) float Sst[DK * LDT];   // state S [dk][dv]
    __shared__ __attribute__((aligned(16))) float Kt [CHUNK * LDT];
    __shared__ __attribute__((aligned(16))) float Qt [CHUNK * LDT];
    __shared__ __attribute__((aligned(16))) float RU [CHUNK * LDT]; // KS0 -> rhs -> U
    __shared__ __attribute__((aligned(16))) float OB [CHUNK * LDT]; // QS0
    __shared__ float Amat[CHUNK * LDA];
    __shared__ float Attm[CHUNK * LDA];
    __shared__ float bvec[CHUNK], evec[CHUNK], betav[CHUNK], dvec[CHUNK], gtmp[CHUNK];

    const int tid  = threadIdx.x;
    const int lane = tid & 31;
    const int wave = tid >> 5;
    const int h    = blockIdx.x;
    const int kh   = h >> 1;               // rep = 2
    const float scale = rsqrtf((float)DK);

    for (int i = tid; i < DK * LDT; i += 256) Sst[i] = 0.f;
    __syncthreads();

    for (int c = 0; c < T_TOKENS / CHUNK; ++c) {
        const int t0 = c * CHUNK;

        // ---- stage K/Q tiles via async global->LDS (raw f32 copy) ----
        {
            const int r  = tid >> 3;              // 0..31
            const int c0 = (tid & 7) * 16;        // 0..112
            const float* xr = x + (size_t)(t0 + r) * QKV_DIM;
#pragma unroll
            for (int i = 0; i < 16; i += 4) {
                async_copy16(xr + KEY_DIM + kh * DK + c0 + i, &Kt[r * LDT + c0 + i]);
                async_copy16(xr + kh * DK + c0 + i,           &Qt[r * LDT + c0 + i]);
            }
        }
        if (tid < CHUNK) {
            gtmp[tid]  = g[(t0 + tid) * NUM_V_HEADS + h];
            betav[tid] = beta[(t0 + tid) * NUM_V_HEADS + h];
        }
        ASYNC_WAIT0();          // this wave's async LDS writes done
        __syncthreads();        // all waves' writes visible
        if (tid == 0) {
            float run = 0.f;
            for (int i = 0; i < CHUNK; ++i) { run += gtmp[i]; bvec[i] = run; }
        }
        __syncthreads();
        if (tid < CHUNK) {
            evec[tid] = __expf(bvec[tid]);                   // e^{b_i}
            dvec[tid] = __expf(bvec[CHUNK - 1] - bvec[tid]); // e^{bL - b_i}
        }

        // ---- G1/G2: RU = K@S0 ; OB = Q@S0  (M=32,N=128,K=128) ----
#pragma unroll
        for (int q = 0; q < 4; ++q) {
            const int tl = wave * 4 + q;
            const float* At = (tl < 16) ? Kt : Qt;
            float*       Dt = (tl < 16) ? RU : OB;
            const int t2 = tl & 15;
            const int mt = t2 >> 3;
            const int nt = t2 & 7;
            v8f acc = {};
#pragma unroll
            for (int ks = 0; ks < 4; ++ks) {
                v16bf a = fragA_f32(At, LDT, mt * 16, ks * 32, lane);
                v16bf b = fragB_kmajor_f32(Sst, LDT, ks * 32, nt * 16, lane);
                acc = WMMA_BF16(a, b, acc);
            }
            storeAcc(Dt, LDT, mt * 16, nt * 16, lane, acc);
        }
        // ---- G3 (waves 0-3): raw K@K^T ; G4 (waves 4-7): raw Q@K^T ----
        {
            const int w4 = wave & 3;
            const int mt = w4 >> 1, nt = w4 & 1;
            const float* At = (wave < 4) ? Kt : Qt;
            float*       Dt = (wave < 4) ? Amat : Attm;
            v8f acc = {};
#pragma unroll
            for (int ks = 0; ks < 4; ++ks) {
                v16bf a = fragA_f32(At, LDT, mt * 16, ks * 32, lane);
                v16bf b = fragB_nmajor_f32(Kt, LDT, ks * 32, nt * 16, lane);
                acc = WMMA_BF16(a, b, acc);
            }
            storeAcc(Dt, LDA, mt * 16, nt * 16, lane, acc);
        }
        __syncthreads();

        // ---- elementwise: decay/mask A,Att ; rhs ; S *= e^{bL} ----
        {
#pragma unroll
            for (int q = 0; q < 4; ++q) {
                const int e = tid + 256 * q;
                const int i = e >> 5, j = e & 31;
                const float dec = __expf(bvec[i] - bvec[j]);
                Amat[i * LDA + j] = (j <  i) ? betav[i] * dec * Amat[i * LDA + j] : 0.f;
                Attm[i * LDA + j] = (j <= i) ? dec * Attm[i * LDA + j] : 0.f;
            }
            const int r  = tid >> 3;
            const int c0 = (tid & 7) * 16;
            const float* vr = x + (size_t)(t0 + r) * QKV_DIM + 2 * KEY_DIM + h * DV + c0;
#pragma unroll
            for (int i = 0; i < 16; ++i)
                RU[r * LDT + c0 + i] = betav[r] * (vr[i] - evec[r] * RU[r * LDT + c0 + i]);

            const float ebL = __expf(bvec[CHUNK - 1]);
            const int sr = tid >> 1, sv = (tid & 1) * 64;
#pragma unroll
            for (int i = 0; i < 64; ++i) Sst[sr * LDT + sv + i] *= ebL;
        }
        __syncthreads();

        // ---- forward substitution: (I+A) U = rhs, U in place in RU ----
        for (int i = 1; i < CHUNK; ++i) {
            if (tid < DV) {
                float s = 0.f;
                for (int j = 0; j < i; ++j) s += Amat[i * LDA + j] * RU[j * LDT + tid];
                RU[i * LDT + tid] -= s;
            }
            __syncthreads();
        }

        // ---- G5: S += (K * e^{bL-b})^T @ U  (M=128,N=128,K=32) ----
        {
            v16bf a = fragAT_scaled_f32(Kt, LDT, wave * 16, dvec, lane);
#pragma unroll
            for (int nt = 0; nt < 8; ++nt) {
                v16bf b = fragB_kmajor_f32(RU, LDT, 0, nt * 16, lane);
                v8f acc = {};
                acc = WMMA_BF16(a, b, acc);
                addAcc(Sst, LDT, wave * 16, nt * 16, lane, acc);
            }
        }
        // ---- G6: O = scale*(e^{b} ⊙ QS0 + Att @ U) -> global ----
#pragma unroll
        for (int q = 0; q < 2; ++q) {
            const int tl = wave * 2 + q;
            const int mt = tl >> 3, nt = tl & 7;
            v16bf a = fragA_f32(Attm, LDA, mt * 16, 0, lane);
            v16bf b = fragB_kmajor_f32(RU, LDT, 0, nt * 16, lane);
            v8f acc = {};
            acc = WMMA_BF16(a, b, acc);
            const int ln = lane & 15, gg = lane >> 4;
#pragma unroll
            for (int i = 0; i < 8; ++i) {
                const int row = mt * 16 + gg * 8 + i;
                const int col = nt * 16 + ln;
                o[(size_t)(t0 + row) * VALUE_DIM + h * DV + col] =
                    scale * (evec[row] * OB[row * LDT + col] + acc[i]);
            }
        }
        __syncthreads();
    }
}

// =====================================================================
// Gated RMSNorm + SiLU(z) gate, in place over o. rows = T*32, wave/row.
// =====================================================================
__global__ __launch_bounds__(256)
void rms_gate_kernel(float* __restrict__ o, const float* __restrict__ z,
                     const float* __restrict__ nw)
{
    const int row  = blockIdx.x * 8 + (threadIdx.x >> 5);
    const int lane = threadIdx.x & 31;
    if (row >= T_TOKENS * NUM_V_HEADS) return;
    float* op = o + (size_t)row * DV;
    const float* zp = z + (size_t)row * DV;

    float v[4]; float ss = 0.f;
#pragma unroll
    for (int i = 0; i < 4; ++i) { v[i] = op[lane + 32 * i]; ss += v[i] * v[i]; }
#pragma unroll
    for (int m = 16; m; m >>= 1) ss += __shfl_xor(ss, m, 32);
    const float r = rsqrtf(ss * (1.f / DV) + 1e-6f);
#pragma unroll
    for (int i = 0; i < 4; ++i) {
        const int c = lane + 32 * i;
        const float zn = zp[c];
        op[c] = v[i] * r * nw[c] * (zn / (1.f + __expf(-zn)));
    }
}

// =====================================================================
// host-side launcher
// =====================================================================
extern "C" void kernel_launch(void* const* d_in, const int* in_sizes, int n_in,
                              void* d_out, int out_size, void* d_ws, size_t ws_size,
                              hipStream_t stream)
{
    (void)in_sizes; (void)n_in; (void)out_size; (void)ws_size;
    const float* hidden  = (const float*)d_in[0];
    const float* W_qkv   = (const float*)d_in[1];
    const float* W_z     = (const float*)d_in[2];
    const float* W_b     = (const float*)d_in[3];
    const float* W_a     = (const float*)d_in[4];
    const float* conv_w  = (const float*)d_in[5];
    const float* dt_bias = (const float*)d_in[6];
    const float* A_log   = (const float*)d_in[7];
    const float* norm_w  = (const float*)d_in[8];
    const float* W_out   = (const float*)d_in[9];
    float* out = (float*)d_out;

    char* ws = (char*)d_ws;
    float* mixed = (float*)ws;                                        // [T, 8192]
    float* o_buf = mixed;                                             // [T, 4096] (alias)
    float* x_buf = (float*)(ws + (size_t)T_TOKENS * QKV_DIM * 4);     // [T, 8192]
    float* z_buf = (float*)(ws + (size_t)T_TOKENS * QKV_DIM * 8);     // [T, 4096]
    float* b_buf = z_buf + (size_t)T_TOKENS * VALUE_DIM;
    float* a_buf = b_buf + T_TOKENS * NUM_V_HEADS;
    float* beta  = a_buf + T_TOKENS * NUM_V_HEADS;
    float* g_buf = beta  + T_TOKENS * NUM_V_HEADS;

    wmma_gemm_nt<<<dim3(QKV_DIM / 32,   T_TOKENS / 64), 256, 0, stream>>>(
        hidden, W_qkv, mixed, T_TOKENS, QKV_DIM, HIDDEN);
    wmma_gemm_nt<<<dim3(VALUE_DIM / 32, T_TOKENS / 64), 256, 0, stream>>>(
        hidden, W_z, z_buf, T_TOKENS, VALUE_DIM, HIDDEN);
    wmma_gemm_nt<<<dim3(1,              T_TOKENS / 64), 256, 0, stream>>>(
        hidden, W_b, b_buf, T_TOKENS, NUM_V_HEADS, HIDDEN);
    wmma_gemm_nt<<<dim3(1,              T_TOKENS / 64), 256, 0, stream>>>(
        hidden, W_a, a_buf, T_TOKENS, NUM_V_HEADS, HIDDEN);

    {
        const long long n = (long long)T_TOKENS * QKV_DIM;
        conv_silu_kernel<<<(unsigned)((n + 255) / 256), 256, 0, stream>>>(mixed, conv_w, x_buf);
    }

    l2norm_qk_kernel<<<(T_TOKENS * 32) / 8, 256, 0, stream>>>(x_buf);

    beta_g_kernel<<<(T_TOKENS * NUM_V_HEADS + 255) / 256, 256, 0, stream>>>(
        b_buf, a_buf, dt_bias, A_log, beta, g_buf);

    // chunked WMMA delta-rule scan (async LDS staging)
    scan_chunked_kernel<<<NUM_V_HEADS, 256, 0, stream>>>(x_buf, beta, g_buf, o_buf);

    rms_gate_kernel<<<(T_TOKENS * NUM_V_HEADS) / 8, 256, 0, stream>>>(o_buf, z_buf, norm_w);

    wmma_gemm_nt<<<dim3(HIDDEN / 32, T_TOKENS / 64), 256, 0, stream>>>(
        o_buf, W_out, out, T_TOKENS, HIDDEN, VALUE_DIM);
}
